// EpisodicMemoryModule_51737176048047
// MI455X (gfx1250) — matrix operand order repaired and checked by reading, not compiled, not run on hardware
//
#include <hip/hip_runtime.h>
#include <hip/hip_bf16.h>

// ---------------------------------------------------------------------------
// EpisodicMemoryModule on gfx1250 (MI455X): bf16 WMMA everywhere GEMM-shaped.
// ---------------------------------------------------------------------------

typedef __attribute__((ext_vector_type(16))) __bf16 v16bf;
typedef __attribute__((ext_vector_type(8)))  float  v8f;

#define BB    128   // batch
#define NF    512   // facts
#define UU    256   // units
#define NH1   50    // hidden of attention MLP
#define KFEAT 1024  // 4*U feature width
#define U2    512   // 2*U (xr|xh)
#define U3    768   // 3*U

__device__ __forceinline__ int lane_id() { return threadIdx.x & 31; }

// |x| via opaque inline asm: gfx1250 isel crashes on SLP-vectorized fabs
// (malformed i32 bitcast of v2f32 under the sign-mask AND), and InstCombine
// canonicalizes fmax(x,-x) / bit-tricks back into fabs. Inline asm keeps the
// abs out of IR entirely (and blocks SLP through it) while emitting the same
// v_and_b32 the backend would have used.
__device__ __forceinline__ float fabs_opaque(float x) {
  float r;
  asm("v_and_b32 %0, 0x7fffffff, %1" : "=v"(r) : "v"(x));
  return r;
}

// A fragment, 16x32 bf16, row-major source with leading dim ld.
// Lanes 0-15: M=lane, K = 0..7 and 16..23 ; lanes 16-31: M=lane-16, K = 8..15 and 24..31.
__device__ __forceinline__ v16bf load_a_frag(const __bf16* src, int ld, int row0, int k0) {
  int lane = lane_id();
  const __bf16* p = src + (size_t)(row0 + (lane & 15)) * ld + k0 + ((lane >> 4) << 3);
  v16bf a;
#pragma unroll
  for (int j = 0; j < 8; ++j) { a[j] = p[j]; a[j + 8] = p[j + 16]; }
  return a;
}

// B fragment, 32x16 bf16, from N-major ("transposed") weights wt[n][k], leading dim ldk.
// Lane n = lane&15 ; lanes 0-15: K = 0..7 and 16..23 ; lanes 16-31: K = 8..15 and 24..31.
__device__ __forceinline__ v16bf load_b_frag(const __bf16* wt, int ldk, int n0, int k0) {
  int lane = lane_id();
  const __bf16* p = wt + (size_t)(n0 + (lane & 15)) * ldk + k0 + ((lane >> 4) << 3);
  v16bf b;
#pragma unroll
  for (int j = 0; j < 8; ++j) { b[j] = p[j]; b[j + 8] = p[j + 16]; }
  return b;
}

__device__ __forceinline__ v8f wmma_bf16(v16bf a, v16bf b, v8f c) {
  return __builtin_amdgcn_wmma_f32_16x16x32_bf16(false, a, false, b, (short)0, c, false, false);
}

// ------------------------------ prep kernels -------------------------------

__global__ void __launch_bounds__(256) cvt_bf16_kernel(const float* __restrict__ in,
                                                       __bf16* __restrict__ out, int n) {
  int i = blockIdx.x * blockDim.x + threadIdx.x;
  if (i < n) out[i] = (__bf16)in[i];
}

// dst[n*K + k] = (n < nvalid) ? src[k*ld + coloff + n] : 0   (N-major bf16 weights)
__global__ void __launch_bounds__(256) pack_wt_kernel(const float* __restrict__ src,
                                                      __bf16* __restrict__ dst,
                                                      int K, int ld, int coloff,
                                                      int ncols, int nvalid) {
  int i = blockIdx.x * blockDim.x + threadIdx.x;
  if (i >= ncols * K) return;
  int n = i / K, k = i - n * K;
  float v = (n < nvalid) ? src[(size_t)k * ld + coloff + n] : 0.0f;
  dst[i] = (__bf16)v;
}

// ------------------------- attention score kernel --------------------------
// scores[b,n] = tanh(feats @ W1 + b1) @ W2 + b2, feats built on the fly.
// One wave per 16-row tile of one batch. Per 32-wide u-chunk we load facts/q/m
// ONCE and emit all 4 feature segments (seg is compile-time -> branch-free).
__global__ void __launch_bounds__(128) scores_kernel(
    const float* __restrict__ facts, const float* __restrict__ question,
    const float* __restrict__ mem, const __bf16* __restrict__ W1t,
    const float* __restrict__ b1, const float* __restrict__ W2,
    const float* __restrict__ b2, float* __restrict__ scores) {
  int wid = blockIdx.x * (blockDim.x >> 5) + (threadIdx.x >> 5);
  int b  = wid >> 5;        // 32 n-tiles per batch row
  int nt = wid & 31;
  int n0 = nt << 4;
  int lane = lane_id();
  int hi = lane >> 4, m = lane & 15;
  const float* frow = facts + ((size_t)b * NF + n0 + m) * UU;
  const float* qrow = question + (size_t)b * UU;
  const float* mrow = mem + (size_t)b * UU;
  v8f acc[4] = {};
  for (int cc = 0; cc < 8; ++cc) {        // 8 chunks of 32 over U=256
    int ub = (cc << 5) + (hi << 3);
    float f[16], q[16], mv[16];
#pragma unroll
    for (int j = 0; j < 16; ++j) {
      int u = ub + (j < 8 ? j : j + 8);
      f[j] = frow[u]; q[j] = qrow[u]; mv[j] = mrow[u];
    }
#pragma unroll
    for (int seg = 0; seg < 4; ++seg) {   // compile-time segment select
      v16bf a;
#pragma unroll
      for (int j = 0; j < 16; ++j) {
        float feat = (seg == 0) ? f[j] * q[j]
                   : (seg == 1) ? f[j] * mv[j]
                   : (seg == 2) ? fabs_opaque(f[j] - q[j])
                                : fabs_opaque(f[j] - mv[j]);
        a[j] = (__bf16)feat;
      }
      int k0 = ((seg << 3) + cc) << 5;    // K position in the 1024-wide feats
#pragma unroll
      for (int t = 0; t < 4; ++t) {
        v16bf bw = load_b_frag(W1t, KFEAT, t << 4, k0);
        acc[t] = wmma_bf16(a, bw, acc[t]);
      }
    }
  }
  // tanh + @W2 reduction across the 16 lanes of each half-wave
  float part[8];
#pragma unroll
  for (int v = 0; v < 8; ++v) part[v] = 0.0f;
#pragma unroll
  for (int t = 0; t < 4; ++t) {
    int col = (t << 4) + m;
    if (col < NH1) {
      float b1v = b1[col], w2v = W2[col];
#pragma unroll
      for (int v = 0; v < 8; ++v) part[v] += tanhf(acc[t][v] + b1v) * w2v;
    }
  }
#pragma unroll
  for (int v = 0; v < 8; ++v) {
    part[v] += __shfl_xor(part[v], 1);
    part[v] += __shfl_xor(part[v], 2);
    part[v] += __shfl_xor(part[v], 4);
    part[v] += __shfl_xor(part[v], 8);
  }
  if (m == 0) {
#pragma unroll
    for (int v = 0; v < 8; ++v)
      scores[(size_t)b * NF + n0 + v + (hi << 3)] = part[v] + b2[0];
  }
}

// ------------------------------- softmax -----------------------------------
__global__ void __launch_bounds__(NF) softmax_kernel(const float* __restrict__ scores,
                                                     float* __restrict__ att) {
  __shared__ float red[NF];
  int b = blockIdx.x, n = threadIdx.x;
  float s = scores[(size_t)b * NF + n];
  red[n] = s;
  __syncthreads();
  for (int off = NF >> 1; off > 0; off >>= 1) {
    if (n < off) red[n] = fmaxf(red[n], red[n + off]);
    __syncthreads();
  }
  float mx = red[0];
  __syncthreads();
  float e = __expf(s - mx);
  red[n] = e;
  __syncthreads();
  for (int off = NF >> 1; off > 0; off >>= 1) {
    if (n < off) red[n] += red[n + off];
    __syncthreads();
  }
  att[(size_t)b * NF + n] = e / red[0];
}

// --------------------- x-projection (facts @ gru_k[:,U:3U]) ----------------
// (B*N, 256) @ (256, 512) -> xproj stored t-major: xproj[t][b][col] in bf16.
__global__ void __launch_bounds__(256) xproj_kernel(
    const __bf16* __restrict__ facts_bf, const __bf16* __restrict__ gkt,
    const float* __restrict__ gru_b, __bf16* __restrict__ xproj) {
  int wid = blockIdx.x * (blockDim.x >> 5) + (threadIdx.x >> 5);
  int mt = wid >> 3;        // 4096 row-tiles
  int g  = wid & 7;         // 8 groups of 64 cols
  int row0 = mt << 4, n0 = g << 6;
  int lane = lane_id(), hi = lane >> 4, nn = lane & 15;
  v8f c[4] = {};
#pragma unroll
  for (int kk = 0; kk < 8; ++kk) {
    v16bf a = load_a_frag(facts_bf, UU, row0, kk << 5);
#pragma unroll
    for (int u = 0; u < 4; ++u) {
      v16bf bw = load_b_frag(gkt, UU, n0 + (u << 4), kk << 5);
      c[u] = wmma_bf16(a, bw, c[u]);
    }
  }
#pragma unroll
  for (int u = 0; u < 4; ++u) {
    int col = n0 + (u << 4) + nn;
    float bias = gru_b[UU + col];
#pragma unroll
    for (int v = 0; v < 8; ++v) {
      int gm = row0 + v + (hi << 3);
      int b = gm >> 9, n = gm & (NF - 1);
      xproj[((size_t)n * BB + b) * U2 + col] = (__bf16)(c[u][v] + bias);
    }
  }
}

// ------------------------- persistent GRU scan -----------------------------
// One workgroup (32 waves) owns the whole (128 x 256) hidden state.
// Wave (mt,g): rows mt*16..+15, cols g*64..+63 ; h kept in f32 registers,
// h / r*h mirrored as bf16 A-operands in LDS (2 x 64 KB of the 320 KB WGP LDS).
__global__ void __launch_bounds__(1024) gru_scan_kernel(
    const __bf16* __restrict__ xproj, const float* __restrict__ att,
    const __bf16* __restrict__ rkrt, const __bf16* __restrict__ rkht,
    float* __restrict__ episode) {
  __shared__ __bf16 h_lds[BB * UU];
  __shared__ __bf16 rh_lds[BB * UU];
  int wave = threadIdx.x >> 5;
  int lane = lane_id(), hi = lane >> 4, nn = lane & 15;
  int mt = wave >> 2, g = wave & 3;
  int row0 = mt << 4, c0 = g << 6;
  v8f h[4] = {};
  for (int i = threadIdx.x; i < BB * UU; i += 1024) h_lds[i] = (__bf16)0.0f;
  __syncthreads();

  for (int t = 0; t < NF; ++t) {
    const __bf16* xrow = xproj + (size_t)t * BB * U2;
    // phase A: s1 = h @ rkr ; r = sigmoid(xr + s1) ; stage r*h (bf16)
    {
      v8f s1[4] = {};
#pragma unroll 2
      for (int kk = 0; kk < 8; ++kk) {
        v16bf a = load_a_frag(h_lds, UU, row0, kk << 5);
#pragma unroll
        for (int u = 0; u < 4; ++u) {
          v16bf bw = load_b_frag(rkrt, UU, c0 + (u << 4), kk << 5);
          s1[u] = wmma_bf16(a, bw, s1[u]);
        }
      }
#pragma unroll
      for (int u = 0; u < 4; ++u) {
        int col = c0 + (u << 4) + nn;
#pragma unroll
        for (int v = 0; v < 8; ++v) {
          int b = row0 + v + (hi << 3);
          float xr = (float)xrow[(size_t)b * U2 + col];
          float r = 1.0f / (1.0f + __expf(-(xr + s1[u][v])));
          rh_lds[b * UU + col] = (__bf16)(r * h[u][v]);
        }
      }
    }
    __syncthreads();
    // phase B: s2 = (r*h) @ rkh ; h = a*tanh(xh+s2) + (1-a)*h
    {
      v8f s2[4] = {};
#pragma unroll 2
      for (int kk = 0; kk < 8; ++kk) {
        v16bf a = load_a_frag(rh_lds, UU, row0, kk << 5);
#pragma unroll
        for (int u = 0; u < 4; ++u) {
          v16bf bw = load_b_frag(rkht, UU, c0 + (u << 4), kk << 5);
          s2[u] = wmma_bf16(a, bw, s2[u]);
        }
      }
      float at[8];
#pragma unroll
      for (int v = 0; v < 8; ++v)
        at[v] = att[(size_t)(row0 + v + (hi << 3)) * NF + t];
#pragma unroll
      for (int u = 0; u < 4; ++u) {
        int col = c0 + (u << 4) + nn;
#pragma unroll
        for (int v = 0; v < 8; ++v) {
          int b = row0 + v + (hi << 3);
          float xh = (float)xrow[(size_t)b * U2 + UU + col];
          float hh = tanhf(xh + s2[u][v]);
          float hn = at[v] * hh + (1.0f - at[v]) * h[u][v];
          h[u][v] = hn;
          h_lds[b * UU + col] = (__bf16)hn;
        }
      }
    }
    __syncthreads();
    if (t + 1 < NF) __builtin_prefetch(xproj + (size_t)(t + 1) * BB * U2, 0, 1);
  }

#pragma unroll
  for (int u = 0; u < 4; ++u) {
    int col = c0 + (u << 4) + nn;
#pragma unroll
    for (int v = 0; v < 8; ++v) {
      int b = row0 + v + (hi << 3);
      episode[(size_t)b * UU + col] = h[u][v];
    }
  }
}

// ------------------------- memory update -----------------------------------
// out = relu(concat([memory, episode, question]) @ Wm + bm)  (128x768)@(768x256)
__global__ void __launch_bounds__(256) memory_update_kernel(
    const float* __restrict__ mem, const float* __restrict__ episode,
    const float* __restrict__ question, const __bf16* __restrict__ Wmt,
    const float* __restrict__ bm, float* __restrict__ out) {
  int wave = threadIdx.x >> 5;
  int lane = lane_id(), hi = lane >> 4, nn = lane & 15;
  int row0 = wave << 4;            // 8 waves -> 128 rows
  int n0 = blockIdx.x << 4;        // 16 blocks -> 256 cols
  int row = row0 + nn;
  v8f c = {};
#pragma unroll
  for (int src3 = 0; src3 < 3; ++src3) {           // memory | episode | question
    const float* srow = (src3 == 0) ? (mem + (size_t)row * UU)
                      : (src3 == 1) ? (episode + (size_t)row * UU)
                                    : (question + (size_t)row * UU);
#pragma unroll
    for (int cc = 0; cc < 8; ++cc) {               // 8 chunks of 32 per segment
      int ub = (cc << 5) + (hi << 3);
      v16bf a;
#pragma unroll
      for (int j = 0; j < 16; ++j) a[j] = (__bf16)srow[ub + (j < 8 ? j : j + 8)];
      int k0 = ((src3 << 3) + cc) << 5;
      v16bf bw = load_b_frag(Wmt, U3, n0, k0);
      c = wmma_bf16(a, bw, c);
    }
  }
  int col = n0 + nn;
  float bias = bm[col];
#pragma unroll
  for (int v = 0; v < 8; ++v) {
    int b = row0 + v + (hi << 3);
    out[(size_t)b * UU + col] = fmaxf(c[v] + bias, 0.0f);
  }
}

// ------------------------------- launcher ----------------------------------

extern "C" void kernel_launch(void* const* d_in, const int* in_sizes, int n_in,
                              void* d_out, int out_size, void* d_ws, size_t ws_size,
                              hipStream_t stream) {
  (void)in_sizes; (void)n_in; (void)out_size; (void)ws_size;
  const float* facts    = (const float*)d_in[0];
  const float* question = (const float*)d_in[1];
  const float* W1       = (const float*)d_in[2];
  const float* b1       = (const float*)d_in[3];
  const float* W2       = (const float*)d_in[4];
  const float* b2       = (const float*)d_in[5];
  const float* gru_k    = (const float*)d_in[6];
  const float* gru_rk   = (const float*)d_in[7];
  const float* gru_b    = (const float*)d_in[8];
  const float* Wm       = (const float*)d_in[9];
  const float* bm       = (const float*)d_in[10];
  float* out = (float*)d_out;

  char* ws = (char*)d_ws;
  size_t off = 0;
  auto alloc = [&](size_t bytes) -> void* {
    void* p = ws + off;
    off += (bytes + 255) & ~(size_t)255;
    return p;
  };
  __bf16* facts_bf = (__bf16*)alloc((size_t)BB * NF * UU * 2);
  __bf16* xproj    = (__bf16*)alloc((size_t)NF * BB * U2 * 2);
  __bf16* W1t      = (__bf16*)alloc((size_t)64 * KFEAT * 2);
  __bf16* gkt      = (__bf16*)alloc((size_t)U2 * UU * 2);
  __bf16* rkrt     = (__bf16*)alloc((size_t)UU * UU * 2);
  __bf16* rkht     = (__bf16*)alloc((size_t)UU * UU * 2);
  __bf16* Wmt      = (__bf16*)alloc((size_t)UU * U3 * 2);
  float*  scores   = (float*)alloc((size_t)BB * NF * 4);
  float*  att      = (float*)alloc((size_t)BB * NF * 4);
  float*  episode  = (float*)alloc((size_t)BB * UU * 4);
  float*  memA     = (float*)alloc((size_t)BB * UU * 4);
  float*  memB     = (float*)alloc((size_t)BB * UU * 4);

  // One-time (per launch) packing: bf16 facts + N-major bf16 weights.
  int nfacts = BB * NF * UU;
  cvt_bf16_kernel<<<(nfacts + 255) / 256, 256, 0, stream>>>(facts, facts_bf, nfacts);
  pack_wt_kernel<<<(64 * KFEAT + 255) / 256, 256, 0, stream>>>(W1, W1t, KFEAT, NH1, 0, 64, NH1);
  pack_wt_kernel<<<(U2 * UU + 255) / 256, 256, 0, stream>>>(gru_k, gkt, UU, U3, UU, U2, U2);
  pack_wt_kernel<<<(UU * UU + 255) / 256, 256, 0, stream>>>(gru_rk, rkrt, UU, U3, UU, UU, UU);
  pack_wt_kernel<<<(UU * UU + 255) / 256, 256, 0, stream>>>(gru_rk, rkht, UU, U3, 2 * UU, UU, UU);
  pack_wt_kernel<<<(UU * U3 + 255) / 256, 256, 0, stream>>>(Wm, Wmt, U3, UU, 0, UU, UU);
  // Step-invariant projection (facts @ gru_k[:,U:3U] + bias), stored t-major.
  xproj_kernel<<<4096, 256, 0, stream>>>(facts_bf, gkt, gru_b, xproj);

  const float* mem_cur = question;           // initial memory = question
  float* mems[3] = {memA, memB, out};
  for (int s = 0; s < 3; ++s) {
    scores_kernel<<<1024, 128, 0, stream>>>(facts, question, mem_cur, W1t, b1, W2, b2, scores);
    softmax_kernel<<<BB, NF, 0, stream>>>(scores, att);
    gru_scan_kernel<<<1, 1024, 0, stream>>>(xproj, att, rkrt, rkht, episode);
    memory_update_kernel<<<16, 256, 0, stream>>>(mem_cur, episode, question, Wmt, bm, mems[s]);
    mem_cur = mems[s];
  }
}
